// FastNN_67594195304883
// MI455X (gfx1250) — matrix operand order repaired and checked by reading, not compiled, not run on hardware
//
#include <hip/hip_runtime.h>
#include <hip/hip_bf16.h>
#include <math.h>

typedef __attribute__((ext_vector_type(16))) _Float16 v16h;
typedef __attribute__((ext_vector_type(8)))  float    v8f;

#define EPSV 1e-5f

struct ConvBN {
    const float* w;   // conv weight
    const float* cb;  // conv bias
    const float* g;   // bn gamma
    const float* b;   // bn beta
    const float* rm;  // bn running mean
    const float* rv;  // bn running var
};

__device__ __forceinline__ v8f wmma_f16(v16h a, v16h b, v8f c) {
    // D = A(16x32 f16) * B(32x16 f16) + C(16x16 f32)
    return __builtin_amdgcn_wmma_f32_16x16x32_f16(
        /*neg_a=*/false, a, /*neg_b=*/false, b,
        /*c_mod=*/(short)0, c, /*reuse_a=*/false, /*reuse_b=*/false);
}

__device__ __forceinline__ void lds_fence() {
    // CDNA5 split wait counters: make intra-wave LDS RAW explicit and stop
    // the compiler from reordering cross-lane LDS traffic.
    asm volatile("s_wait_dscnt 0" ::: "memory");
}

// ---------------------------------------------------------------------------
// Block activity: active1 from mask1 (4x4 mean > .25), active2 from
// maxpool2(mask1) (4x4 mean of 2x2-maxes > .25). mask values are exactly 0/1.
// ---------------------------------------------------------------------------
__global__ void prep_active(const float* __restrict__ mask1,
                            unsigned char* __restrict__ act1,
                            unsigned char* __restrict__ act2) {
    const int NB1 = 32 * 56 * 56;
    const int NB2 = 32 * 28 * 28;
    int idx = blockIdx.x * blockDim.x + threadIdx.x;
    if (idx < NB1) {
        int n = idx / (56 * 56), r = idx % (56 * 56);
        int gy = r / 56, gx = r % 56;
        const float* m = mask1 + (size_t)n * 224 * 224;
        float s = 0.f;
        for (int i = 0; i < 4; ++i)
            for (int j = 0; j < 4; ++j)
                s += m[(gy * 4 + i) * 224 + gx * 4 + j];
        act1[idx] = (s > 4.0f) ? 1 : 0;                  // mean > 0.25
    } else if (idx < NB1 + NB2) {
        int q = idx - NB1;
        int n = q / (28 * 28), r = q % (28 * 28);
        int gy = r / 28, gx = r % 28;
        const float* m = mask1 + (size_t)n * 224 * 224;
        float s = 0.f;
        for (int i = 0; i < 4; ++i)
            for (int j = 0; j < 4; ++j) {
                int y0 = (gy * 4 + i) * 2, x0 = (gx * 4 + j) * 2;
                float a = m[y0 * 224 + x0],     bq = m[y0 * 224 + x0 + 1];
                float c = m[(y0 + 1) * 224 + x0], d = m[(y0 + 1) * 224 + x0 + 1];
                s += fmaxf(fmaxf(a, bq), fmaxf(c, d));
            }
        act2[q] = (s > 4.0f) ? 1 : 0;
    }
}

// ---------------------------------------------------------------------------
// Pack per-stage WMMA B-fragments + folded (bias, scale, shift) tables.
// B layout (16-bit, 32Kx16N): lane = 16*h + n ; element e -> k = (e<8 ? e : e+8) + 8h.
// Fragment order: conv1 [NH], conv3x3 [9 taps][NH], conv3 [1].
// Tables: b1/s1/h1 (2C each) | b2/s2/h2 (2C each) | b3/s3/h3 (C each).
// ---------------------------------------------------------------------------
template <int C>
__global__ void pack_stage(_Float16* __restrict__ frags, float* __restrict__ tabs,
                           ConvBN c1, ConvBN c2, ConvBN c3) {
    constexpr int TC = 2 * C;
    constexpr int NH = TC / 16;
    constexpr int NFRAG = NH + 9 * NH + 1;
    const int nFragElems = NFRAG * 512;
    int idx = blockIdx.x * blockDim.x + threadIdx.x;
    if (idx < nFragElems) {
        int f = idx >> 9;
        int r = idx & 511;
        int lane = r >> 4;
        int e = r & 15;
        int k = (e < 8 ? e : e + 8) + ((lane & 16) ? 8 : 0);
        int n = lane & 15;
        float v = 0.f;
        if (f < NH) {                                     // conv1: W[TC][C]
            int o = n + 16 * f;
            if (k < C) v = c1.w[o * C + k];
        } else if (f < NH + 9 * NH) {                     // conv3x3: W[TC][TC][3][3]
            int gq = f - NH;
            int tap = gq / NH, nh = gq % NH;
            int o = n + 16 * nh;
            if (k < TC) v = c2.w[(o * TC + k) * 9 + tap];
        } else {                                          // conv3: W[C][TC]
            if (n < C && k < TC) v = c3.w[n * TC + k];
        }
        frags[idx] = (_Float16)v;
    } else {
        int t = idx - nFragElems;
        if (t < 15 * C) {
            const ConvBN* p; int t2 = t; int width;
            if (t2 < 3 * TC)       { p = &c1; width = TC; }
            else if (t2 < 6 * TC)  { p = &c2; t2 -= 3 * TC; width = TC; }
            else                   { p = &c3; t2 -= 6 * TC; width = C; }
            int kind = t2 / width, ch = t2 % width;
            float sc = p->g[ch] * rsqrtf(p->rv[ch] + EPSV);
            float v = (kind == 0) ? p->cb[ch]
                    : (kind == 1) ? sc
                                  : (p->b[ch] - p->rm[ch] * sc);
            tabs[t] = v;
        }
    }
}

// ---------------------------------------------------------------------------
// Dense pre-stage conv1x1 + ReLU + BN.  Stage1: f32 x (cin=3) -> f16 y (16ch).
// ---------------------------------------------------------------------------
__global__ void pre1(const float* __restrict__ x, _Float16* __restrict__ y1, ConvBN p) {
    const int HW = 224 * 224;
    int idx = blockIdx.x * blockDim.x + threadIdx.x;
    if (idx >= 32 * HW) return;
    int n = idx / HW, r = idx % HW;
    const float* xp = x + (size_t)n * 3 * HW + r;
    float x0 = xp[0], x1 = xp[(size_t)HW], x2 = xp[(size_t)2 * HW];
    _Float16* yp = y1 + (size_t)n * 16 * HW + r;
    for (int co = 0; co < 16; ++co) {
        float z = p.cb[co] + x0 * p.w[co * 3] + x1 * p.w[co * 3 + 1] + x2 * p.w[co * 3 + 2];
        float sc = p.g[co] * rsqrtf(p.rv[co] + EPSV);
        float v = sc * fmaxf(z, 0.f) + (p.b[co] - p.rm[co] * sc);
        yp[(size_t)co * HW] = (_Float16)v;
    }
}

// Stage2: f16 x2 (cin=16) -> f16 y2 (8ch), 112x112.
__global__ void pre2(const _Float16* __restrict__ xin, _Float16* __restrict__ yout, ConvBN p) {
    const int HW = 112 * 112;
    int idx = blockIdx.x * blockDim.x + threadIdx.x;
    if (idx >= 32 * HW) return;
    int n = idx / HW, r = idx % HW;
    const _Float16* xp = xin + (size_t)n * 16 * HW + r;
    float xi[16];
    for (int ci = 0; ci < 16; ++ci) xi[ci] = (float)xp[(size_t)ci * HW];
    _Float16* yp = yout + (size_t)n * 8 * HW + r;
    for (int co = 0; co < 8; ++co) {
        float z = p.cb[co];
        for (int ci = 0; ci < 16; ++ci) z += xi[ci] * p.w[co * 16 + ci];
        float sc = p.g[co] * rsqrtf(p.rv[co] + EPSV);
        float v = sc * fmaxf(z, 0.f) + (p.b[co] - p.rm[co] * sc);
        yp[(size_t)co * HW] = (_Float16)v;
    }
}

// ---------------------------------------------------------------------------
// 2x2 maxpool on f16 [NC, Hin, Win].
// ---------------------------------------------------------------------------
__global__ void pool2(const _Float16* __restrict__ in, _Float16* __restrict__ out,
                      int NC, int Hin, int Win) {
    int Ho = Hin >> 1, Wo = Win >> 1;
    int idx = blockIdx.x * blockDim.x + threadIdx.x;
    if (idx >= NC * Ho * Wo) return;
    int c = idx / (Ho * Wo), r = idx % (Ho * Wo);
    int ho = r / Wo, wo = r % Wo;
    const _Float16* ip = in + (size_t)c * Hin * Win + (size_t)(ho * 2) * Win + wo * 2;
    float a = (float)ip[0], b = (float)ip[1];
    float d = (float)ip[Win], e = (float)ip[Win + 1];
    out[idx] = (_Float16)fmaxf(fmaxf(a, b), fmaxf(d, e));
}

// ---------------------------------------------------------------------------
// Sparse block path. One wave == one 4x4 block.  All GEMMs via
// v_wmma_f32_16x16x32_f16 with M = 16 pixels.  In-place scatter: only active
// blocks are rewritten, so inactive blocks keep the dense-pre value (== the
// reference's jnp.where(pix, d, y)).
// ---------------------------------------------------------------------------
template <int C>
__global__ __launch_bounds__(256) void block_kernel(
    _Float16* __restrict__ ymap, const unsigned char* __restrict__ act,
    const _Float16* __restrict__ frags, const float* __restrict__ tabs,
    int GH, int GW, int Hm, int Wm) {
    constexpr int TC = 2 * C;
    constexpr int NH = TC / 16;
    constexpr int WAVES = 8;
    __shared__ _Float16 sA[WAVES][TC][16];   // y block, later h2
    __shared__ _Float16 sB[WAVES][TC][16];   // h1

    int wave = threadIdx.x >> 5;
    int lane = threadIdx.x & 31;
    int blk  = blockIdx.x * WAVES + wave;
    int total = 32 * GH * GW;
    if (blk >= total) return;        // wave-uniform
    if (!act[blk]) return;           // wave-uniform: EXEC all-ones below

    int n = blk / (GH * GW), r = blk % (GH * GW);
    int gy = r / GW, gx = r % GW;
    size_t plane = (size_t)Hm * Wm;
    const _Float16* yb = ymap + (size_t)n * C * plane + (size_t)(gy * 4) * Wm + gx * 4;

    // --- load 4x4 x C block into LDS: sA[c][pixel] ---
    for (int t = lane; t < C * 16; t += 32) {
        int c = t >> 4, p = t & 15;
        sA[wave][c][p] = yb[(size_t)c * plane + (p >> 2) * Wm + (p & 3)];
    }
    lds_fence();

    const v16h* F = (const v16h*)frags;
    const int m  = lane & 15;              // pixel row of this lane's A half
    const int hi = (lane & 16) ? 8 : 0;    // k-offset for upper lane half
    const int mhi = (lane & 16) ? 8 : 0;   // m-offset for D rows

    // --- conv1: 16x(K=C pad32) x TC ---
    v16h a;
#pragma unroll
    for (int e = 0; e < 16; ++e) {
        int k = (e < 8 ? e : e + 8) + hi;
        int kc = (k < C) ? k : 0;
        _Float16 v = (_Float16)0.f;
        if (k < C) v = sA[wave][kc][m];
        a[e] = v;
    }
    const v8f zero8 = {0.f, 0.f, 0.f, 0.f, 0.f, 0.f, 0.f, 0.f};
#pragma unroll
    for (int nh = 0; nh < NH; ++nh) {
        v16h b = F[(size_t)nh * 32 + lane];
        v8f c = wmma_f16(a, b, zero8);
        int ch = (lane & 15) + 16 * nh;
        float bias = tabs[ch], sc = tabs[TC + ch], sh = tabs[2 * TC + ch];
#pragma unroll
        for (int rr = 0; rr < 8; ++rr) {
            float v = sc * fmaxf(c[rr] + bias, 0.f) + sh;
            sB[wave][ch][rr + mhi] = (_Float16)v;
        }
    }
    lds_fence();

    // --- conv3x3 within the block (per-block zero padding): 9 im2col taps ---
    v8f c2[NH];
#pragma unroll
    for (int nh = 0; nh < NH; ++nh) c2[nh] = zero8;
#pragma unroll
    for (int tap = 0; tap < 9; ++tap) {
        int dy = tap / 3 - 1, dx = tap % 3 - 1;
        v16h at;
#pragma unroll
        for (int e = 0; e < 16; ++e) {
            int k = (e < 8 ? e : e + 8) + hi;
            int py = (m >> 2) + dy, px = (m & 3) + dx;
            bool ok = (k < TC) && (py >= 0) && (py < 4) && (px >= 0) && (px < 4);
            int kc = (k < TC) ? k : 0;
            int q  = ok ? (py * 4 + px) : 0;
            _Float16 v = (_Float16)0.f;
            if (ok) v = sB[wave][kc][q];
            at[e] = v;
        }
#pragma unroll
        for (int nh = 0; nh < NH; ++nh) {
            v16h b = F[(size_t)(NH + tap * NH + nh) * 32 + lane];
            c2[nh] = wmma_f16(at, b, c2[nh]);
        }
    }
#pragma unroll
    for (int nh = 0; nh < NH; ++nh) {
        int ch = (lane & 15) + 16 * nh;
        float bias = tabs[3 * TC + ch], sc = tabs[4 * TC + ch], sh = tabs[5 * TC + ch];
#pragma unroll
        for (int rr = 0; rr < 8; ++rr) {
            float v = sc * fmaxf(c2[nh][rr] + bias, 0.f) + sh;
            sA[wave][ch][rr + mhi] = (_Float16)v;   // h2 (y block fully consumed)
        }
    }
    lds_fence();

    // --- conv3: 16x(K=TC pad32) x C ---
    v16h a3;
#pragma unroll
    for (int e = 0; e < 16; ++e) {
        int k = (e < 8 ? e : e + 8) + hi;
        int kc = (k < TC) ? k : 0;
        _Float16 v = (_Float16)0.f;
        if (k < TC) v = sA[wave][kc][m];
        a3[e] = v;
    }
    v16h b3 = F[(size_t)(10 * NH) * 32 + lane];
    v8f c3 = wmma_f16(a3, b3, zero8);

    // --- scatter: overwrite the active block in ymap ---
    if ((lane & 15) < C) {
        int ch = lane & 15;
        float bias = tabs[6 * TC + ch], sc = tabs[6 * TC + C + ch], sh = tabs[6 * TC + 2 * C + ch];
        _Float16* op = ymap + (size_t)n * C * plane + (size_t)ch * plane +
                       (size_t)(gy * 4) * Wm + gx * 4;
#pragma unroll
        for (int rr = 0; rr < 8; ++rr) {
            int mm = rr + mhi;
            float v = sc * fmaxf(c3[rr] + bias, 0.f) + sh;
            op[(mm >> 2) * Wm + (mm & 3)] = (_Float16)v;
        }
    }
}

// ---------------------------------------------------------------------------
// FC (10 x 25088) + softmax.  One workgroup per image.
// ---------------------------------------------------------------------------
__global__ __launch_bounds__(256) void fc_softmax(const _Float16* __restrict__ x3,
                                                  const float* __restrict__ fw,
                                                  const float* __restrict__ fb,
                                                  float* __restrict__ out) {
    __shared__ float red[256];
    __shared__ float sl[10];
    const int FDIM = 8 * 56 * 56;  // 25088
    int n = blockIdx.x, tid = threadIdx.x;
    const _Float16* xp = x3 + (size_t)n * FDIM;
    float acc[10];
#pragma unroll
    for (int c = 0; c < 10; ++c) acc[c] = 0.f;
    for (int i = tid; i < FDIM; i += 256) {
        float xv = (float)xp[i];
#pragma unroll
        for (int c = 0; c < 10; ++c) acc[c] += xv * fw[(size_t)c * FDIM + i];
    }
    for (int c = 0; c < 10; ++c) {
        red[tid] = acc[c];
        __syncthreads();
        for (int s = 128; s > 0; s >>= 1) {
            if (tid < s) red[tid] += red[tid + s];
            __syncthreads();
        }
        if (tid == 0) sl[c] = red[0] + fb[c];
        __syncthreads();
    }
    if (tid == 0) {
        float mx = sl[0];
        for (int c = 1; c < 10; ++c) mx = fmaxf(mx, sl[c]);
        float s = 0.f, e[10];
        for (int c = 0; c < 10; ++c) { e[c] = expf(sl[c] - mx); s += e[c]; }
        for (int c = 0; c < 10; ++c) out[n * 10 + c] = e[c] / s;
    }
}

// ---------------------------------------------------------------------------
// Launch.  d_in flattening follows setup_inputs() dict insertion order:
// x, mask1, params{s1{cw,cb,cbn{g,b,rm,rv},w1,b1,bn1{..},w2,b2,bn2{..},
// w3,b3,bn3{..}}, s2{same}, fc_w, fc_b}
// ---------------------------------------------------------------------------
extern "C" void kernel_launch(void* const* d_in, const int* in_sizes, int n_in,
                              void* d_out, int out_size, void* d_ws, size_t ws_size,
                              hipStream_t stream) {
    (void)in_sizes; (void)n_in; (void)out_size; (void)ws_size;
    auto P = [&](int i) { return (const float*)d_in[i]; };
    const float* x     = P(0);
    const float* mask1 = P(1);
    ConvBN s1c  {P(2),  P(3),  P(4),  P(5),  P(6),  P(7)};
    ConvBN s1c1 {P(8),  P(9),  P(10), P(11), P(12), P(13)};
    ConvBN s1c2 {P(14), P(15), P(16), P(17), P(18), P(19)};
    ConvBN s1c3 {P(20), P(21), P(22), P(23), P(24), P(25)};
    ConvBN s2c  {P(26), P(27), P(28), P(29), P(30), P(31)};
    ConvBN s2c1 {P(32), P(33), P(34), P(35), P(36), P(37)};
    ConvBN s2c2 {P(38), P(39), P(40), P(41), P(42), P(43)};
    ConvBN s2c3 {P(44), P(45), P(46), P(47), P(48), P(49)};
    const float* fcw = P(50);
    const float* fcb = P(51);

    char* ws = (char*)d_ws;
    size_t off = 0;
    auto A = [&](size_t bytes) -> char* {
        char* p = ws + off;
        off = (off + bytes + 255) & ~(size_t)255;
        return p;
    };
    _Float16* y1 = (_Float16*)A((size_t)32 * 16 * 224 * 224 * 2);
    _Float16* x2 = (_Float16*)A((size_t)32 * 16 * 112 * 112 * 2);
    _Float16* y2 = (_Float16*)A((size_t)32 * 8 * 112 * 112 * 2);
    _Float16* x3 = (_Float16*)A((size_t)32 * 8 * 56 * 56 * 2);
    unsigned char* act1 = (unsigned char*)A(32 * 56 * 56);
    unsigned char* act2 = (unsigned char*)A(32 * 28 * 28);
    _Float16* wf1 = (_Float16*)A(21 * 512 * 2);
    _Float16* wf2 = (_Float16*)A(11 * 512 * 2);
    float* tb1 = (float*)A(240 * 4);
    float* tb2 = (float*)A(120 * 4);

    const int NB1 = 32 * 56 * 56, NB2 = 32 * 28 * 28;
    prep_active<<<(NB1 + NB2 + 255) / 256, 256, 0, stream>>>(mask1, act1, act2);
    pack_stage<16><<<(21 * 512 + 240 + 255) / 256, 256, 0, stream>>>(wf1, tb1, s1c1, s1c2, s1c3);
    pack_stage<8><<<(11 * 512 + 120 + 255) / 256, 256, 0, stream>>>(wf2, tb2, s2c1, s2c2, s2c3);

    pre1<<<(32 * 224 * 224 + 255) / 256, 256, 0, stream>>>(x, y1, s1c);
    block_kernel<16><<<NB1 / 8, 256, 0, stream>>>(y1, act1, wf1, tb1, 56, 56, 224, 224);
    pool2<<<(32 * 16 * 112 * 112 + 255) / 256, 256, 0, stream>>>(y1, x2, 32 * 16, 224, 224);

    pre2<<<(32 * 112 * 112 + 255) / 256, 256, 0, stream>>>(x2, y2, s2c);
    block_kernel<8><<<NB2 / 8, 256, 0, stream>>>(y2, act2, wf2, tb2, 28, 28, 112, 112);
    pool2<<<(32 * 8 * 56 * 56 + 255) / 256, 256, 0, stream>>>(y2, x3, 32 * 8, 112, 112);

    fc_softmax<<<32, 256, 0, stream>>>(x3, fcw, fcb, (float*)d_out);
}
